// InverseDynamicsGNN_31714038513930
// MI455X (gfx1250) — matrix operand order repaired and checked by reading, not compiled, not run on hardware
//
#include <hip/hip_runtime.h>
#include <cstdint>
#include <cstddef>

// ---------------- problem constants ----------------
#define HIDW   256
#define BATCH  256
#define NNODE  64
#define NEDGE  512
#define TOKN   (NNODE*BATCH)     // 16384 node tokens
#define TOKE   (NEDGE*BATCH)     // 131072 edge tokens
#define NCHUNK (TOKE/TOKN)       // 8 chunks of 16384 edge-token rows
#define NSV    133
#define GNN_ITERS 6
#define LN_EPS 1e-5f

typedef _Float16 half_t;
typedef __attribute__((ext_vector_type(16))) half_t v16h;
typedef __attribute__((ext_vector_type(8)))  float  v8f;

union Frag { v16h v; uint32_t u[8]; };
union HPack { half_t h[2]; uint32_t u; };

// ---------------- fragment loaders (wave32 WMMA 16x16x32 f16 layouts) -------
// A (16x32): lane l holds row M=l&15; k = (l>>4)*8 + (v&3)*2 + (v>>2)*16 + h
__device__ __forceinline__ v16h load_a_rm(const half_t* A, int row, int K, int kt, int lane) {
  Frag f;
  const half_t* base = A + (size_t)row * K + kt * 32;
  const int khalf = (lane >> 4) * 8;
  #pragma unroll
  for (int v = 0; v < 8; ++v) {
    const int k0 = khalf + (v & 3) * 2 + ((v >> 2) * 16);
    f.u[v] = *reinterpret_cast<const uint32_t*>(base + k0);
  }
  return f.v;
}
// B (32x16) fragment out of the LDS-staged packed weights
__device__ __forceinline__ v16h load_b_lds(const uint32_t* ldsb, int nt, int lane) {
  Frag f;
  const uint32_t* p = ldsb + nt * 256 + lane * 8;
  #pragma unroll
  for (int v = 0; v < 8; ++v) f.u[v] = p[v];
  return f.v;
}

__device__ __forceinline__ float redux16(float x) {  // sum within 16-lane group
  x += __shfl_xor(x, 1, 32);
  x += __shfl_xor(x, 2, 32);
  x += __shfl_xor(x, 4, 32);
  x += __shfl_xor(x, 8, 32);
  return x;
}

// ---------------- fused GEMM (+bias,+LayerNorm,+ReLU) -> f16 [rows x 256] ---
// MODE 0: A = A0 row-major [rows x (KT*32)]
// MODE 1: message net. KT=3. k-tiles 0..1 gather x[edge_src[e]] (K=64),
//         k-tile 2 reads Fbuf (per-edge-token constants, K=32) at global row.
// MODE 2: update net. KT=4. k-tiles 0..1 from A0 (m_hat, K=64),
//         k-tiles 2..3 from A1 (x, K=64).
template<int KT, int MODE>
__global__ __launch_bounds__(256, 1)
void gemm_ln_relu(const half_t* __restrict__ A0,
                  const half_t* __restrict__ A1,
                  const half_t* __restrict__ Fbuf,
                  const int*    __restrict__ esrc,
                  int rowOff,
                  const uint32_t* __restrict__ Wp,
                  const float* __restrict__ bias,
                  const float* __restrict__ gamma,
                  const float* __restrict__ beta,
                  half_t* __restrict__ Out) {
  constexpr int NT = 16;                       // 256 output cols
  __shared__ uint32_t ldsb[NT * 256];          // one k-step of packed B (16 KB)
  const int lane   = threadIdx.x & 31;
  const int wave   = threadIdx.x >> 5;
  const int mTile  = blockIdx.x * 8 + wave;
  const int rowBase = mTile * 16;
  const int mrow    = rowBase + (lane & 15);   // A-row (chunk-local)

  int xrow = 0;
  if (MODE == 1) {
    const int e    = (rowOff + rowBase) >> 8;              // tile within one edge
    const int srcn = esrc[e];
    const int bidx = (rowBase & 255) + (lane & 15);
    xrow = srcn * BATCH + bidx;
  }

  v8f acc[NT];
  #pragma unroll
  for (int nt = 0; nt < NT; ++nt) acc[nt] = (v8f)0.0f;

  for (int kt = 0; kt < KT; ++kt) {
    // --- cooperative stage of this k-step's 16 B fragments into LDS ---
    {
      const uint4* s4 = reinterpret_cast<const uint4*>(Wp + (size_t)kt * NT * 256);
      uint4* d4 = reinterpret_cast<uint4*>(ldsb);
      #pragma unroll
      for (int j = 0; j < NT / 4; ++j) d4[j * 256 + threadIdx.x] = s4[j * 256 + threadIdx.x];
    }
    __syncthreads();

    v16h a;
    if (MODE == 0) {
      a = load_a_rm(A0, mrow, KT * 32, kt, lane);
    } else if (MODE == 1) {
      if (kt < 2) a = load_a_rm(A0, xrow, 64, kt, lane);
      else        a = load_a_rm(Fbuf, rowOff + mrow, 32, 0, lane);
    } else { // MODE 2
      if (kt < 2) a = load_a_rm(A0, mrow, 64, kt, lane);
      else        a = load_a_rm(A1, mrow, 64, kt - 2, lane);
    }
    #pragma unroll
    for (int nt = 0; nt < NT; ++nt) {
      v16h b = load_b_lds(ldsb, nt, lane);
      acc[nt] = __builtin_amdgcn_wmma_f32_16x16x32_f16(
          false, a, false, b, (short)0, acc[nt], false, false);
    }
    __syncthreads();
  }

  const int colL = lane & 15;
  const int rsel = lane >> 4;

  // bias (per-column)
  #pragma unroll
  for (int nt = 0; nt < NT; ++nt) {
    const float bv = bias[nt * 16 + colL];
    #pragma unroll
    for (int r = 0; r < 8; ++r) acc[nt][r] += bv;
  }

  // LayerNorm over 256 cols per row (row M = r + 8*rsel in this lane's group)
  float mean[8], rstd[8];
  #pragma unroll
  for (int r = 0; r < 8; ++r) {
    float s = 0.f;
    #pragma unroll
    for (int nt = 0; nt < NT; ++nt) s += acc[nt][r];
    mean[r] = redux16(s) * (1.0f / 256.0f);
    float q = 0.f;
    #pragma unroll
    for (int nt = 0; nt < NT; ++nt) {
      const float d = acc[nt][r] - mean[r];
      q += d * d;
    }
    rstd[r] = rsqrtf(redux16(q) * (1.0f / 256.0f) + LN_EPS);
  }

  #pragma unroll
  for (int nt = 0; nt < NT; ++nt) {
    const float g  = gamma[nt * 16 + colL];
    const float be = beta[nt * 16 + colL];
    #pragma unroll
    for (int r = 0; r < 8; ++r) {
      float vv = (acc[nt][r] - mean[r]) * rstd[r] * g + be;
      vv = fmaxf(vv, 0.0f);
      const int row = rowBase + r + 8 * rsel;
      Out[(size_t)row * 256 + nt * 16 + colL] = (half_t)vv;
    }
  }
}

// ---------------- final 256->64 layer ---------------------------------------
// STYLE 0: plain, store f16 [rows x 64]  (input / update nets)
// STYLE 1: tanh, then atomic scatter-add into m_hat via edge_dst (message net)
template<int STYLE>
__global__ __launch_bounds__(256, 1)
void gemm_out64(const half_t* __restrict__ A,
                const uint32_t* __restrict__ Wp,
                const float* __restrict__ bias,
                int rowOff,
                const int* __restrict__ edst,
                float* __restrict__ mhat,
                half_t* __restrict__ OutH) {
  constexpr int NT = 4, KT = 8;               // 64 cols, K = 256
  __shared__ uint32_t ldsb[NT * 256];         // 4 KB
  const int lane    = threadIdx.x & 31;
  const int wave    = threadIdx.x >> 5;
  const int mTile   = blockIdx.x * 8 + wave;
  const int rowBase = mTile * 16;
  const int mrow    = rowBase + (lane & 15);

  v8f acc[NT];
  #pragma unroll
  for (int nt = 0; nt < NT; ++nt) acc[nt] = (v8f)0.0f;

  for (int kt = 0; kt < KT; ++kt) {
    {
      const uint4* s4 = reinterpret_cast<const uint4*>(Wp + (size_t)kt * NT * 256);
      uint4* d4 = reinterpret_cast<uint4*>(ldsb);
      if (threadIdx.x < NT * 64) d4[threadIdx.x] = s4[threadIdx.x];
    }
    __syncthreads();
    v16h a = load_a_rm(A, mrow, 256, kt, lane);
    #pragma unroll
    for (int nt = 0; nt < NT; ++nt) {
      v16h b = load_b_lds(ldsb, nt, lane);
      acc[nt] = __builtin_amdgcn_wmma_f32_16x16x32_f16(
          false, a, false, b, (short)0, acc[nt], false, false);
    }
    __syncthreads();
  }

  const int colL = lane & 15;
  const int rsel = lane >> 4;
  int dstn = 0;
  if (STYLE == 1) dstn = edst[(rowOff + rowBase) >> 8];

  #pragma unroll
  for (int nt = 0; nt < NT; ++nt) {
    const float bv = bias[nt * 16 + colL];
    #pragma unroll
    for (int r = 0; r < 8; ++r) {
      float vv = acc[nt][r] + bv;
      const int row = rowBase + r + 8 * rsel;     // chunk-local
      const int col = nt * 16 + colL;
      if (STYLE == 1) {
        vv = tanhf(vv);
        const int b = row & 255;                  // rowOff is a multiple of 256
        atomicAdd(&mhat[((size_t)dstn * BATCH + b) * 64 + col], vv);
      } else {
        OutH[(size_t)row * 64 + col] = (half_t)vv;
      }
    }
  }
}

// ---------------- prep / small kernels --------------------------------------
__global__ void pack_w(const float* __restrict__ W, uint32_t* __restrict__ dst,
                       int din, int dout, int KT) {
  const int id = blockIdx.x * 256 + threadIdx.x;
  const int NT = dout >> 4;
  if (id >= KT * NT * 256) return;
  const int v    = id & 7;
  const int lane = (id >> 3) & 31;
  const int nt   = (id >> 8) % NT;
  const int kt   = (id >> 8) / NT;
  const int k0   = kt * 32 + (lane >> 4) * 8 + (v & 3) * 2 + (v >> 2) * 16;
  const int col  = nt * 16 + (lane & 15);
  HPack p;
  p.h[0] = (half_t)((k0     < din) ? W[(size_t)k0       * dout + col] : 0.f);
  p.h[1] = (half_t)((k0 + 1 < din) ? W[(size_t)(k0 + 1) * dout + col] : 0.f);
  dst[id] = p.u;
}

__global__ void build_node_in(const float* __restrict__ state,
                              const float* __restrict__ node_feat,
                              half_t* __restrict__ NodeIn) {
  const int t = blockIdx.x * 256 + threadIdx.x;
  if (t >= TOKN) return;
  const int n = t >> 8, b = t & 255;
  half_t* o = NodeIn + (size_t)t * 32;
  const float* st = state + (size_t)b * (2 * NSV);
  #pragma unroll
  for (int j = 0; j < 6; ++j)  o[j]      = (half_t)node_feat[n * 6 + j];
  #pragma unroll
  for (int j = 0; j < 5; ++j)  o[6 + j]  = (half_t)st[j];
  #pragma unroll
  for (int j = 0; j < 5; ++j)  o[11 + j] = (half_t)st[NSV + j];
  o[16] = (half_t)st[5 + n];
  o[17] = (half_t)st[5 + NNODE + n];
  o[18] = (half_t)st[NSV + 5 + n];
  o[19] = (half_t)st[NSV + 5 + NNODE + n];
  #pragma unroll
  for (int j = 20; j < 32; ++j) o[j] = (half_t)0.f;
}

__global__ void build_F(const half_t* __restrict__ NodeIn,
                        const float* __restrict__ ef,
                        const int* __restrict__ esrc,
                        half_t* __restrict__ Fbuf) {
  const int t = blockIdx.x * 256 + threadIdx.x;
  if (t >= TOKE) return;
  const int e = t >> 8, b = t & 255;
  half_t* o = Fbuf + (size_t)t * 32;
  const half_t* src = NodeIn + ((size_t)esrc[e] * BATCH + b) * 32;
  o[0] = (half_t)ef[e];
  #pragma unroll
  for (int j = 0; j < 20; ++j) o[1 + j] = src[j];
  #pragma unroll
  for (int j = 21; j < 32; ++j) o[j] = (half_t)0.f;
}

__global__ void calc_invc(const int* __restrict__ edst, float* __restrict__ invc) {
  const int n = threadIdx.x;
  if (n >= NNODE) return;
  int c = 0;
  for (int e = 0; e < NEDGE; ++e) c += (edst[e] == n) ? 1 : 0;
  invc[n] = 1.0f / fmaxf((float)c, 1.0f);
}

__global__ void zero_f32(float* __restrict__ p, int n) {
  const int i = blockIdx.x * 256 + threadIdx.x;
  if (i < n) p[i] = 0.f;
}

__global__ void mhat_to_h16(const float* __restrict__ mhat,
                            const float* __restrict__ invc,
                            half_t* __restrict__ MhatH) {
  const int i = blockIdx.x * 256 + threadIdx.x;
  if (i >= TOKN * 64) return;
  const int tok = i >> 6;
  MhatH[i] = (half_t)(mhat[i] * invc[tok >> 8]);
}

// output heads: actions[b,n] = tanh(h.Wout+bo); sigmoids[b] += sigmoid(h.Ws+bs)/64
__global__ void head_kernel(const half_t* __restrict__ H2,
                            const float* __restrict__ Wout, const float* __restrict__ bout,
                            const float* __restrict__ Wsig, const float* __restrict__ bsig,
                            float* __restrict__ out) {
  const int t    = blockIdx.x * 8 + (threadIdx.x >> 5);
  const int lane = threadIdx.x & 31;
  float so = 0.f, ss = 0.f;
  #pragma unroll
  for (int j = 0; j < 8; ++j) {
    const float h = (float)H2[(size_t)t * 256 + lane * 8 + j];
    so += h * Wout[lane * 8 + j];
    ss += h * Wsig[lane * 8 + j];
  }
  #pragma unroll
  for (int m = 1; m < 32; m <<= 1) { so += __shfl_xor(so, m, 32); ss += __shfl_xor(ss, m, 32); }
  if (lane == 0) {
    const int n = t >> 8, b = t & 255;
    out[(size_t)b * NNODE + n] = tanhf(so + bout[0]);
    const float sg = 1.0f / (1.0f + expf(-(ss + bsig[0])));
    atomicAdd(&out[TOKN + b], sg * (1.0f / (float)NNODE));
  }
}

// ---------------- launcher ---------------------------------------------------
extern "C" void kernel_launch(void* const* d_in, const int* in_sizes, int n_in,
                              void* d_out, int out_size, void* d_ws, size_t ws_size,
                              hipStream_t stream) {
  (void)in_sizes; (void)n_in; (void)out_size; (void)ws_size;
  const float* state     = (const float*)d_in[0];
  const float* node_feat = (const float*)d_in[1];
  const float* edge_feat = (const float*)d_in[2];
  const int*   esrc      = (const int*)d_in[3];
  const int*   edst      = (const int*)d_in[4];
  // params flatten: per net {l1 W,b, ln1 g,b, l2 W,b, ln2 g,b, out W,b (,sig W,b)}
  const float* P[42];
  for (int i = 0; i < 42; ++i) P[i] = (const float*)d_in[5 + i];
  #define IN_(k)  P[0  + (k)]
  #define MSG(k)  P[10 + (k)]
  #define UPD(k)  P[20 + (k)]
  #define OUTP(k) P[30 + (k)]
  float* outp = (float*)d_out;

  // workspace layout
  char* ws = (char*)d_ws;
  size_t off = 0;
  auto alloc = [&](size_t bytes) { void* p = ws + off; off = (off + bytes + 255) & ~(size_t)255; return p; };
  uint32_t* wp_in1 = (uint32_t*)alloc((size_t)32  * 256 * 2);
  uint32_t* wp_in2 = (uint32_t*)alloc((size_t)256 * 256 * 2);
  uint32_t* wp_in3 = (uint32_t*)alloc((size_t)256 * 64  * 2);
  uint32_t* wp_m1  = (uint32_t*)alloc((size_t)96  * 256 * 2);
  uint32_t* wp_m2  = (uint32_t*)alloc((size_t)256 * 256 * 2);
  uint32_t* wp_m3  = (uint32_t*)alloc((size_t)256 * 64  * 2);
  uint32_t* wp_u1  = (uint32_t*)alloc((size_t)128 * 256 * 2);
  uint32_t* wp_u2  = (uint32_t*)alloc((size_t)256 * 256 * 2);
  uint32_t* wp_u3  = (uint32_t*)alloc((size_t)256 * 64  * 2);
  uint32_t* wp_o1  = (uint32_t*)alloc((size_t)64  * 256 * 2);
  uint32_t* wp_o2  = (uint32_t*)alloc((size_t)256 * 256 * 2);
  half_t* NodeIn = (half_t*)alloc((size_t)TOKN * 32 * 2);
  half_t* Fbuf   = (half_t*)alloc((size_t)TOKE * 32 * 2);
  half_t* Xbuf   = (half_t*)alloc((size_t)TOKN * 64 * 2);
  half_t* H1     = (half_t*)alloc((size_t)TOKN * 256 * 2);
  half_t* H2     = (half_t*)alloc((size_t)TOKN * 256 * 2);
  float*  mhat   = (float*) alloc((size_t)TOKN * 64 * 4);
  half_t* MhatH  = (half_t*)alloc((size_t)TOKN * 64 * 2);
  float*  invc   = (float*) alloc((size_t)NNODE * 4);

  const dim3 blk(256);
  auto packGrid = [](int KT, int dout) { return dim3((KT * (dout >> 4) * 256 + 255) / 256); };

  // --- pack all weights into B-fragment layout (once per call) ---
  pack_w<<<packGrid(1, 256), blk, 0, stream>>>(IN_(0), wp_in1, 20,  256, 1);
  pack_w<<<packGrid(8, 256), blk, 0, stream>>>(IN_(4), wp_in2, 256, 256, 8);
  pack_w<<<packGrid(8, 64),  blk, 0, stream>>>(IN_(8), wp_in3, 256, 64,  8);
  pack_w<<<packGrid(3, 256), blk, 0, stream>>>(MSG(0), wp_m1,  85,  256, 3);
  pack_w<<<packGrid(8, 256), blk, 0, stream>>>(MSG(4), wp_m2,  256, 256, 8);
  pack_w<<<packGrid(8, 64),  blk, 0, stream>>>(MSG(8), wp_m3,  256, 64,  8);
  pack_w<<<packGrid(4, 256), blk, 0, stream>>>(UPD(0), wp_u1,  128, 256, 4);
  pack_w<<<packGrid(8, 256), blk, 0, stream>>>(UPD(4), wp_u2,  256, 256, 8);
  pack_w<<<packGrid(8, 64),  blk, 0, stream>>>(UPD(8), wp_u3,  256, 64,  8);
  pack_w<<<packGrid(2, 256), blk, 0, stream>>>(OUTP(0), wp_o1, 64,  256, 2);
  pack_w<<<packGrid(8, 256), blk, 0, stream>>>(OUTP(4), wp_o2, 256, 256, 8);

  // --- features ---
  build_node_in<<<dim3(TOKN / 256), blk, 0, stream>>>(state, node_feat, NodeIn);
  build_F<<<dim3(TOKE / 256), blk, 0, stream>>>(NodeIn, edge_feat, esrc, Fbuf);
  calc_invc<<<dim3(1), dim3(64), 0, stream>>>(edst, invc);

  const dim3 gemmGrid(TOKN / 16 / 8);   // 1024 M-tiles, 8 waves/block

  // --- input net: NodeIn(32) -> H1 -> H2 -> x(64) ---
  gemm_ln_relu<1, 0><<<gemmGrid, blk, 0, stream>>>(NodeIn, nullptr, nullptr, nullptr, 0,
      wp_in1, IN_(1), IN_(2), IN_(3), H1);
  gemm_ln_relu<8, 0><<<gemmGrid, blk, 0, stream>>>(H1, nullptr, nullptr, nullptr, 0,
      wp_in2, IN_(5), IN_(6), IN_(7), H2);
  gemm_out64<0><<<gemmGrid, blk, 0, stream>>>(H2, wp_in3, IN_(9), 0, nullptr, nullptr, Xbuf);

  // --- message passing iterations ---
  for (int it = 0; it < GNN_ITERS; ++it) {
    zero_f32<<<dim3(TOKN * 64 / 256), blk, 0, stream>>>(mhat, TOKN * 64);
    for (int c = 0; c < NCHUNK; ++c) {
      const int rowOff = c * TOKN;
      gemm_ln_relu<3, 1><<<gemmGrid, blk, 0, stream>>>(Xbuf, nullptr, Fbuf, esrc, rowOff,
          wp_m1, MSG(1), MSG(2), MSG(3), H1);
      gemm_ln_relu<8, 0><<<gemmGrid, blk, 0, stream>>>(H1, nullptr, nullptr, nullptr, 0,
          wp_m2, MSG(5), MSG(6), MSG(7), H2);
      gemm_out64<1><<<gemmGrid, blk, 0, stream>>>(H2, wp_m3, MSG(9), rowOff, edst, mhat, nullptr);
    }
    mhat_to_h16<<<dim3(TOKN * 64 / 256), blk, 0, stream>>>(mhat, invc, MhatH);
    gemm_ln_relu<4, 2><<<gemmGrid, blk, 0, stream>>>(MhatH, Xbuf, nullptr, nullptr, 0,
        wp_u1, UPD(1), UPD(2), UPD(3), H1);
    gemm_ln_relu<8, 0><<<gemmGrid, blk, 0, stream>>>(H1, nullptr, nullptr, nullptr, 0,
        wp_u2, UPD(5), UPD(6), UPD(7), H2);
    gemm_out64<0><<<gemmGrid, blk, 0, stream>>>(H2, wp_u3, UPD(9), 0, nullptr, nullptr, Xbuf);
  }

  // --- output net + heads ---
  gemm_ln_relu<2, 0><<<gemmGrid, blk, 0, stream>>>(Xbuf, nullptr, nullptr, nullptr, 0,
      wp_o1, OUTP(1), OUTP(2), OUTP(3), H1);
  gemm_ln_relu<8, 0><<<gemmGrid, blk, 0, stream>>>(H1, nullptr, nullptr, nullptr, 0,
      wp_o2, OUTP(5), OUTP(6), OUTP(7), H2);
  zero_f32<<<dim3(1), blk, 0, stream>>>(outp + TOKN, BATCH);
  head_kernel<<<dim3(TOKN / 8), blk, 0, stream>>>(H2, OUTP(8), OUTP(9), OUTP(10), OUTP(11), outp);
}
